// LSTM_37409165148589
// MI455X (gfx1250) — matrix operand order repaired and checked by reading, not compiled, not run on hardware
//
#include <hip/hip_runtime.h>
#include <hip/hip_bf16.h>
#include <stdint.h>

// ---------------------------------------------------------------------------
// Types for CDNA5 WMMA (gfx1250, wave32)
// ---------------------------------------------------------------------------
typedef __bf16 v16bf __attribute__((ext_vector_type(16)));
typedef float  v8f   __attribute__((ext_vector_type(8)));

#define T_SEQ 8192
#define HID_  1024
#define G4H   4096      // 4*HID
#define EMB_  300
#define EMBP  320       // EMB padded to multiple of 32 for WMMA K
#define RELU_ 512
#define REC_BLOCKS 64   // persistent recurrence blocks (64 x 128KB LDS == whole Whh)

// ---------------------------------------------------------------------------
// fp32 -> bf16 conversion / layout kernels (one-time per call, tiny)
// ---------------------------------------------------------------------------
__global__ void cvt_pad_rows_k(const float* __restrict__ src, __bf16* __restrict__ dst,
                               int rows, int cols, int colsPad) {
    size_t total = (size_t)rows * colsPad;
    for (size_t i = (size_t)blockIdx.x * blockDim.x + threadIdx.x; i < total;
         i += (size_t)gridDim.x * blockDim.x) {
        int r = (int)(i / colsPad);
        int c = (int)(i % colsPad);
        dst[i] = (c < cols) ? (__bf16)src[(size_t)r * cols + c] : (__bf16)0.0f;
    }
}

// src is [N,K] row-major (PyTorch weight), dst is [Kpad,N] bf16 (WMMA B layout)
__global__ void cvt_transpose_k(const float* __restrict__ src, __bf16* __restrict__ dst,
                                int N_, int K_, int Kpad) {
    size_t total = (size_t)Kpad * N_;
    for (size_t i = (size_t)blockIdx.x * blockDim.x + threadIdx.x; i < total;
         i += (size_t)gridDim.x * blockDim.x) {
        int k = (int)(i / N_);
        int n = (int)(i % N_);
        dst[i] = (k < K_) ? (__bf16)src[(size_t)n * K_ + k] : (__bf16)0.0f;
    }
}

__global__ void cvt_copy_k(const float* __restrict__ src, __bf16* __restrict__ dst, size_t n) {
    for (size_t i = (size_t)blockIdx.x * blockDim.x + threadIdx.x; i < n;
         i += (size_t)gridDim.x * blockDim.x)
        dst[i] = (__bf16)src[i];
}

// ---------------------------------------------------------------------------
// bf16 WMMA GEMM: C[M,N] (fp32) = A[M,K](bf16,row) x B[K,N](bf16,row) + bias0+bias1
// Block = 256 threads = 8 waves; each wave computes a 64(M) x 16(N) strip.
// Block tile: 64M x 128N. Requires M%64==0, N%128==0, K%32==0.
// ---------------------------------------------------------------------------
__global__ __launch_bounds__(256)
void gemm_bf16_wmma_k(const __bf16* __restrict__ A, const __bf16* __restrict__ B,
                      const float* __restrict__ bias0, const float* __restrict__ bias1,
                      float* __restrict__ C, int M, int N, int K,
                      int lda, int ldb, int ldc) {
    const int wave = threadIdx.x >> 5;
    const int lane = threadIdx.x & 31;
    const int n0    = blockIdx.x * 128 + wave * 16;
    const int mBase = blockIdx.y * 64;

    const v8f vzero = {0.f, 0.f, 0.f, 0.f, 0.f, 0.f, 0.f, 0.f};
    v8f acc[4] = {vzero, vzero, vzero, vzero};

    const int am = lane & 15;        // A fragment: row within tile
    const int kh = lane >> 4;        // A fragment: K-half select
    const int bk = lane;             // unused; B fragment column is n0 + (lane & 15)

    union BV { uint4 q[2]; v16bf v; };

    for (int k0 = 0; k0 < K; k0 += 32) {
        // ---- B fragment: lane L holds col (n0+L%16), K = kh*8 + 0..7 and 16 + kh*8 + 0..7
        const __bf16* pB = B + (size_t)(k0 + kh * 8) * ldb + n0 + am;
        BV ub;
        #pragma unroll
        for (int e = 0; e < 8; ++e) {
            ub.v[e]     = pB[(size_t)e * ldb];
            ub.v[8 + e] = pB[(size_t)(16 + e) * ldb];
        }
        __builtin_prefetch(pB + (size_t)32 * ldb, 0, 1);   // global_prefetch_b8

        // ---- 4 stacked A fragments, 4 WMMAs reusing the B fragment
        #pragma unroll
        for (int s = 0; s < 4; ++s) {
            const __bf16* pA = A + (size_t)(mBase + s * 16 + am) * lda + k0 + kh * 8;
            BV ua;
            ua.q[0] = *(const uint4*)(pA);        // K = kh*8 + 0..7
            ua.q[1] = *(const uint4*)(pA + 16);   // K = 16 + kh*8 + 0..7
            acc[s] = __builtin_amdgcn_wmma_f32_16x16x32_bf16(
                false, ua.v, false, ub.v, (short)0, acc[s], false, false);
        }
    }

    // ---- epilogue: C layout n=lane&15, m = r + 8*(lane>>4); fold both biases
    const int cn = lane & 15;
    const int mh = lane >> 4;
    const float bv = bias0[n0 + cn] + bias1[n0 + cn];
    #pragma unroll
    for (int s = 0; s < 4; ++s) {
        #pragma unroll
        for (int r = 0; r < 8; ++r) {
            C[(size_t)(mBase + s * 16 + mh * 8 + r) * ldc + n0 + cn] = acc[s][r] + bv;
        }
    }
}

// ---------------------------------------------------------------------------
// Persistent LSTM recurrence. 64 blocks x 256 threads; block b owns hidden
// units [16b, 16b+16). Its 64 Whh rows (4 gates x 16 units, bf16, 128 KB)
// live in LDS for all 8192 steps. Per step: 64 length-1024 dots (4 lanes per
// row, shfl reduce), gate nonlinearities, h published to a double-buffered
// global vector, monotonic atomic grid barrier.
// ---------------------------------------------------------------------------
__device__ __forceinline__ float bf16bits_to_f(uint32_t bits16) {
    union { uint32_t u; float f; } cv;
    cv.u = bits16 << 16;
    return cv.f;
}
__device__ __forceinline__ float sigmoidf_(float x) {
    return 1.0f / (1.0f + __expf(-x));
}

__global__ __launch_bounds__(256)
void lstm_rec_k(const __bf16* __restrict__ Whh_b, const float* __restrict__ xg,
                __bf16* __restrict__ hbuf /* 2 x 1024 bf16, double buffered */,
                __bf16* __restrict__ hseq /* [T,1024] or nullptr */,
                float* __restrict__ hlast /* [1024] or nullptr */,
                unsigned* __restrict__ sync, int nsteps) {
    extern __shared__ char smem[];
    uint32_t* wlds  = (uint32_t*)smem;                     // 64 rows x 512 uint (2 bf16) = 128 KB
    float*    hs    = (float*)(smem + 64 * 512 * 4);       // 1024 floats
    float*    gates = (float*)(smem + 64 * 512 * 4 + 1024 * 4); // 64 floats

    const int tid = threadIdx.x;
    const int j0  = blockIdx.x * 16;          // first owned hidden unit

    // ---- stage this block's 64 Whh rows (row r = gate(r>>4), unit(r&15)) into LDS
    for (int i = tid; i < 64 * 512; i += 256) {
        int r  = i >> 9;
        int kk = i & 511;
        int g  = r >> 4, u = r & 15;
        const __bf16* src = Whh_b + (size_t)(g * HID_ + j0 + u) * HID_ + kk * 2;
        union { __bf16 h[2]; uint32_t u32; } cv;
        cv.h[0] = src[0];
        cv.h[1] = src[1];
        wlds[i] = cv.u32;
    }
    __syncthreads();

    float c = 0.0f;                 // cell state: thread tid<16 owns unit j0+tid
    const int row  = tid >> 2;      // 0..63
    const int part = tid & 3;       // quarter of the K dimension

    for (int t = 0; t < nsteps; ++t) {
        // ---- broadcast h_{t-1} into LDS as fp32 (reads buffer t&1; t==0 -> zeros)
        const __bf16* hsrc = hbuf + (size_t)(t & 1) * HID_;
        for (int i = tid; i < HID_; i += 256)
            hs[i] = (t == 0) ? 0.0f : (float)hsrc[i];
        __syncthreads();

        // ---- 64 dot products, 4 lanes each (256 MACs/lane)
        float sum = 0.0f;
        const uint32_t* wr = &wlds[row * 512 + part * 128];
        const float*    hp = &hs[part * 256];
        #pragma unroll 8
        for (int kk = 0; kk < 128; ++kk) {
            uint32_t w2 = wr[kk];
            sum = fmaf(bf16bits_to_f(w2 & 0xffffu), hp[2 * kk], sum);
            sum = fmaf(bf16bits_to_f(w2 >> 16),     hp[2 * kk + 1], sum);
        }
        sum += __shfl_down(sum, 2, 4);
        sum += __shfl_down(sum, 1, 4);
        if (part == 0) {
            int g = row >> 4, u = row & 15;
            gates[row] = sum + xg[(size_t)t * G4H + g * HID_ + j0 + u];
        }
        __syncthreads();

        // ---- cell update for the 16 owned units
        if (tid < 16) {
            float gi = sigmoidf_(gates[tid]);
            float gf = sigmoidf_(gates[16 + tid]);
            float gg = tanhf(gates[32 + tid]);
            float go = sigmoidf_(gates[48 + tid]);
            c = gf * c + gi * gg;
            float h = go * tanhf(c);
            __bf16 hb = (__bf16)h;
            hbuf[(size_t)((t + 1) & 1) * HID_ + j0 + tid] = hb;
            if (hseq)  hseq[(size_t)t * HID_ + j0 + tid] = hb;
            if (hlast && t == nsteps - 1) hlast[j0 + tid] = h;
        }
        __syncthreads();

        // ---- grid barrier: monotonic counter, release/acquire at agent scope
        if (tid == 0) {
            __builtin_amdgcn_fence(__ATOMIC_RELEASE, "agent");
            atomicAdd(sync, 1u);
            unsigned target = (unsigned)gridDim.x * (unsigned)(t + 1);
            while (__hip_atomic_load(sync, __ATOMIC_RELAXED, __HIP_MEMORY_SCOPE_AGENT) < target)
                __builtin_amdgcn_s_sleep(1);
            __builtin_amdgcn_fence(__ATOMIC_ACQUIRE, "agent");
        }
        __syncthreads();
    }
}

// ---------------------------------------------------------------------------
// MLP head + log_softmax (tiny: one block)
// ---------------------------------------------------------------------------
__global__ __launch_bounds__(256)
void mlp_head_k(const float* __restrict__ h2,
                const float* __restrict__ W0, const float* __restrict__ b0,
                const float* __restrict__ W1, const float* __restrict__ b1,
                const float* __restrict__ W2, const float* __restrict__ b2,
                float* __restrict__ out) {
    __shared__ float x1[RELU_];
    __shared__ float x2[RELU_];
    const int tid = threadIdx.x;

    for (int o = tid; o < RELU_; o += 256) {
        float s = b0[o];
        const float* w = W0 + (size_t)o * HID_;
        for (int k = 0; k < HID_; ++k) s = fmaf(w[k], h2[k], s);
        x1[o] = fmaxf(s, 0.0f);
    }
    __syncthreads();
    for (int o = tid; o < RELU_; o += 256) {
        float s = b1[o];
        const float* w = W1 + (size_t)o * RELU_;
        for (int k = 0; k < RELU_; ++k) s = fmaf(w[k], x1[k], s);
        x2[o] = fmaxf(s, 0.0f);
    }
    __syncthreads();
    if (tid == 0) {
        float lg[3];
        for (int o = 0; o < 3; ++o) {
            float s = b2[o];
            const float* w = W2 + (size_t)o * RELU_;
            for (int k = 0; k < RELU_; ++k) s = fmaf(w[k], x2[k], s);
            lg[o] = s;
        }
        float m = fmaxf(lg[0], fmaxf(lg[1], lg[2]));
        float e = expf(lg[0] - m) + expf(lg[1] - m) + expf(lg[2] - m);
        float lse = m + logf(e);
        out[0] = lg[0] - lse;
        out[1] = lg[1] - lse;
        out[2] = lg[2] - lse;
    }
}

// ---------------------------------------------------------------------------
// Host launcher
// ---------------------------------------------------------------------------
extern "C" void kernel_launch(void* const* d_in, const int* in_sizes, int n_in,
                              void* d_out, int out_size, void* d_ws, size_t ws_size,
                              hipStream_t stream) {
    (void)in_sizes; (void)n_in; (void)out_size; (void)ws_size;
    const float* quote = (const float*)d_in[0];
    const float* Wih0  = (const float*)d_in[1];
    const float* Whh0  = (const float*)d_in[2];
    const float* bih0  = (const float*)d_in[3];
    const float* bhh0  = (const float*)d_in[4];
    const float* Wih1  = (const float*)d_in[5];
    const float* Whh1  = (const float*)d_in[6];
    const float* bih1  = (const float*)d_in[7];
    const float* bhh1  = (const float*)d_in[8];
    const float* W0    = (const float*)d_in[9];
    const float* b0    = (const float*)d_in[10];
    const float* W1    = (const float*)d_in[11];
    const float* b1    = (const float*)d_in[12];
    const float* W2    = (const float*)d_in[13];
    const float* b2    = (const float*)d_in[14];

    // ---- workspace layout (~184 MB)
    char* ws = (char*)d_ws;
    size_t off = 0;
    auto carve = [&](size_t bytes) -> void* {
        off = (off + 255) & ~(size_t)255;
        void* p = ws + off;
        off += bytes;
        return p;
    };
    __bf16*   A0    = (__bf16*)  carve((size_t)T_SEQ * EMBP * 2);     // quote, padded bf16
    __bf16*   B0T   = (__bf16*)  carve((size_t)EMBP * G4H * 2);       // Wih0^T bf16
    __bf16*   B1T   = (__bf16*)  carve((size_t)HID_ * G4H * 2);       // Wih1^T bf16
    __bf16*   Whh0b = (__bf16*)  carve((size_t)G4H * HID_ * 2);
    __bf16*   Whh1b = (__bf16*)  carve((size_t)G4H * HID_ * 2);
    float*    xg    = (float*)   carve((size_t)T_SEQ * G4H * 4);      // reused by both layers
    __bf16*   h1mat = (__bf16*)  carve((size_t)T_SEQ * HID_ * 2);
    __bf16*   hbuf0 = (__bf16*)  carve((size_t)2 * HID_ * 2);
    __bf16*   hbuf1 = (__bf16*)  carve((size_t)2 * HID_ * 2);
    float*    h2fin = (float*)   carve((size_t)HID_ * 4);
    unsigned* sync0 = (unsigned*)carve(256);
    unsigned* sync1 = (unsigned*)carve(256);

    hipMemsetAsync(sync0, 0, 512, stream);   // both counters

    // ---- bf16 conversions / layouts
    cvt_pad_rows_k<<<4096, 256, 0, stream>>>(quote, A0, T_SEQ, EMB_, EMBP);
    cvt_transpose_k<<<4096, 256, 0, stream>>>(Wih0, B0T, G4H, EMB_, EMBP);
    cvt_transpose_k<<<8192, 256, 0, stream>>>(Wih1, B1T, G4H, HID_, HID_);
    cvt_copy_k<<<8192, 256, 0, stream>>>(Whh0, Whh0b, (size_t)G4H * HID_);
    cvt_copy_k<<<8192, 256, 0, stream>>>(Whh1, Whh1b, (size_t)G4H * HID_);

    const size_t recLds = (size_t)64 * 512 * 4 + 1024 * 4 + 64 * 4;   // 135424 B

    // ---- layer 0: xg0 = quote @ Wih0^T + (bih0+bhh0), then recurrence -> h1
    gemm_bf16_wmma_k<<<dim3(G4H / 128, T_SEQ / 64), 256, 0, stream>>>(
        A0, B0T, bih0, bhh0, xg, T_SEQ, G4H, EMBP, EMBP, G4H, G4H);
    lstm_rec_k<<<REC_BLOCKS, 256, recLds, stream>>>(
        Whh0b, xg, hbuf0, h1mat, nullptr, sync0, T_SEQ);

    // ---- layer 1: xg1 = h1 @ Wih1^T + (bih1+bhh1), then recurrence -> final h
    gemm_bf16_wmma_k<<<dim3(G4H / 128, T_SEQ / 64), 256, 0, stream>>>(
        h1mat, B1T, bih1, bhh1, xg, T_SEQ, G4H, HID_, HID_, G4H, G4H);
    lstm_rec_k<<<REC_BLOCKS, 256, recLds, stream>>>(
        Whh1b, xg, hbuf1, nullptr, h2fin, sync1, T_SEQ);

    // ---- head
    mlp_head_k<<<1, 256, 0, stream>>>(h2fin, W0, b0, W1, b1, W2, b2, (float*)d_out);
}
